// Neighboorhood_Embedding_81260781240661
// MI455X (gfx1250) — matrix operand order repaired and checked by reading, not compiled, not run on hardware
//
#include <hip/hip_runtime.h>
#include <hip/hip_bf16.h>
#include <cstdint>
#include <cstddef>

// ---------------------------------------------------------------------------
// PCT neighborhood embedding for MI455X (gfx1250, wave32, WMMA).
// GEMMs use v_wmma_f32_16x16x32_f16 (probe-confirmed builtin).
// A-frag layout (16-bit A 16x32): lane L (h=L>>4, m=L&15), VGPR v holds
//   K = (v&4?16:0) + h*8 + (v&3)*2 .. +1  -> two contiguous 16B LDS loads.
// B-frag layout (16-bit B 32x16): lane L (h=L>>4, n=L&15), VGPR v holds
//   K = (v&4?16:0) + h*8 + (v&3)*2 .. +1  -> two contiguous 16B global loads.
// C/D layout: lane L (h,n), VGPR r -> element (m = r + 8*h, n).
// ---------------------------------------------------------------------------

#define EPSV 1e-5f

typedef __attribute__((ext_vector_type(16))) _Float16 v16h;
typedef __attribute__((ext_vector_type(8)))  float    v8f;
typedef __attribute__((ext_vector_type(4)))  unsigned uv4;

union F16Frag { v16h v; uv4 q[2]; unsigned u[8]; _Float16 h[16]; };
union HPair   { unsigned u; _Float16 h[2]; };

// ---------------- fold BN scale into weights (f16), bias stays f32 ---------
__global__ void fold_bn_kernel(const float* __restrict__ w,
                               const float* __restrict__ g,
                               const float* __restrict__ b,
                               const float* __restrict__ m,
                               const float* __restrict__ v,
                               int cout, int cin,
                               _Float16* __restrict__ wh,
                               float* __restrict__ bias) {
  int t = blockIdx.x * blockDim.x + threadIdx.x;
  int stride = gridDim.x * blockDim.x;
  int total = cout * cin;
  for (int i = t; i < total; i += stride) {
    int o = i / cin;
    float s = g[o] * rsqrtf(v[o] + EPSV);
    wh[i] = (_Float16)(w[i] * s);
  }
  for (int o = t; o < cout; o += stride) {
    float s = g[o] * rsqrtf(v[o] + EPSV);
    bias[o] = b[o] - m[o] * s;
  }
}

// ---------------- input MLP: [B*N,3] -> relu(bn(W1 x)) -> relu(bn(W2 .)) ---
__global__ __launch_bounds__(128) void input_embed_kernel(
    const float* __restrict__ x,
    const _Float16* __restrict__ w0, const float* __restrict__ b0,   // [64,3]
    const _Float16* __restrict__ w1, const float* __restrict__ b1,   // [64,64]
    _Float16* __restrict__ h0, int total) {
  __shared__ _Float16 sw0[64 * 3];
  __shared__ _Float16 sw1[64 * 64];
  __shared__ float sb0[64], sb1[64];
  const int tid = threadIdx.x;
  for (int i = tid; i < 64 * 3; i += 128) sw0[i] = w0[i];
  for (int i = tid; i < 64 * 64; i += 128) sw1[i] = w1[i];
  if (tid < 64) { sb0[tid] = b0[tid]; sb1[tid] = b1[tid]; }
  __syncthreads();
  const int p = blockIdx.x * 128 + tid;
  if (p >= total) return;
  const float px = x[(size_t)p * 3 + 0];
  const float py = x[(size_t)p * 3 + 1];
  const float pz = x[(size_t)p * 3 + 2];
  float h[64];
#pragma unroll
  for (int o = 0; o < 64; ++o) {
    float a = sb0[o] + px * (float)sw0[o * 3 + 0]
                     + py * (float)sw0[o * 3 + 1]
                     + pz * (float)sw0[o * 3 + 2];
    h[o] = fmaxf(a, 0.f);
  }
  for (int o = 0; o < 64; ++o) {
    float a = sb1[o];
#pragma unroll
    for (int c = 0; c < 64; ++c) a += h[c] * (float)sw1[o * 64 + c];
    h0[(size_t)p * 64 + o] = (_Float16)fmaxf(a, 0.f);
  }
}

// ---------------- farthest point sampling: 1 workgroup / batch -------------
__global__ __launch_bounds__(1024) void fps_kernel(
    const float* __restrict__ coords,   // [B,N,3]
    int N, int S,
    int* __restrict__ sidx,             // [B,S]
    float* __restrict__ scoords) {      // [B,S,3]
  __shared__ float dist[4096];
  __shared__ unsigned long long redkey, redwin;
  const int b = blockIdx.x;
  const int tid = threadIdx.x;
  const int lane = tid & 31;
  const float* cb = coords + (size_t)b * N * 3;
  const float p0x = cb[0], p0y = cb[1], p0z = cb[2];
  for (int i = tid; i < N; i += 1024) {
    float dx = cb[i * 3 + 0] - p0x;
    float dy = cb[i * 3 + 1] - p0y;
    float dz = cb[i * 3 + 2] - p0z;
    dist[i] = dx * dx + dy * dy + dz * dz;
  }
  if (tid == 0) {
    redkey = 0ULL;
    sidx[(size_t)b * S] = 0;
    scoords[(size_t)b * S * 3 + 0] = p0x;
    scoords[(size_t)b * S * 3 + 1] = p0y;
    scoords[(size_t)b * S * 3 + 2] = p0z;
  }
  __syncthreads();
  for (int s = 1; s < S; ++s) {
    // local argmax (tie -> smaller index, matching jnp.argmax)
    float bd = -1.f; int bi = 0x7FFFFFFF;
    for (int i = tid; i < N; i += 1024) {
      float d = dist[i];
      if (d > bd || (d == bd && i < bi)) { bd = d; bi = i; }
    }
#pragma unroll
    for (int msk = 16; msk > 0; msk >>= 1) {
      float od = __shfl_xor(bd, msk, 32);
      int   oi = __shfl_xor(bi, msk, 32);
      if (od > bd || (od == bd && oi < bi)) { bd = od; bi = oi; }
    }
    if (lane == 0) {
      unsigned long long key =
          ((unsigned long long)__float_as_uint(bd) << 32) |
          (unsigned long long)(0xFFFFFFFFu - (unsigned)bi);
      atomicMax(&redkey, key);
    }
    __syncthreads();
    if (tid == 0) { redwin = redkey; redkey = 0ULL; }
    __syncthreads();
    const int w = (int)(0xFFFFFFFFu - (unsigned)(redwin & 0xFFFFFFFFull));
    const float wx = cb[w * 3 + 0], wy = cb[w * 3 + 1], wz = cb[w * 3 + 2];
    if (tid == 0) {
      sidx[(size_t)b * S + s] = w;
      scoords[((size_t)b * S + s) * 3 + 0] = wx;
      scoords[((size_t)b * S + s) * 3 + 1] = wy;
      scoords[((size_t)b * S + s) * 3 + 2] = wz;
    }
    for (int i = tid; i < N; i += 1024) {
      float dx = cb[i * 3 + 0] - wx;
      float dy = cb[i * 3 + 1] - wy;
      float dz = cb[i * 3 + 2] - wz;
      float d = dx * dx + dy * dy + dz * dz;
      dist[i] = fminf(dist[i], d);
    }
    __syncthreads();
  }
}

// ---------------- kNN (k=32, ascending distance): 1 thread / query ---------
#define KNN_K 32
__global__ __launch_bounds__(64) void knn_kernel(
    const float* __restrict__ q,   // [B,S,3]
    const float* __restrict__ p,   // [B,N,3]
    int N, int S, int B_,
    int* __restrict__ nidx) {      // [B,S,32]
  __shared__ float bd[64][KNN_K];
  __shared__ int   bi[64][KNN_K];
  const int t = blockIdx.x * 64 + threadIdx.x;
  if (t >= B_ * S) return;
  const int b = t / S;
  const float qx = q[(size_t)t * 3 + 0];
  const float qy = q[(size_t)t * 3 + 1];
  const float qz = q[(size_t)t * 3 + 2];
  float* md = bd[threadIdx.x];
  int*   mi = bi[threadIdx.x];
#pragma unroll
  for (int j = 0; j < KNN_K; ++j) { md[j] = 3.4e38f; mi[j] = 0x7FFFFFFF; }
  float curmax = 3.4e38f; int curpos = 0;
  const float* pb = p + (size_t)b * N * 3;
  for (int i = 0; i < N; ++i) {
    float dx = pb[i * 3 + 0] - qx;
    float dy = pb[i * 3 + 1] - qy;
    float dz = pb[i * 3 + 2] - qz;
    float d = dx * dx + dy * dy + dz * dz;
    if (d < curmax) {
      md[curpos] = d; mi[curpos] = i;
      float nm = -1.f; int np = 0;
#pragma unroll
      for (int j = 0; j < KNN_K; ++j)
        if (md[j] > nm) { nm = md[j]; np = j; }
      curmax = nm; curpos = np;
    }
  }
  int* outp = nidx + (size_t)t * KNN_K;
  for (int a = 0; a < KNN_K; ++a) {
    float bestd = 3.5e38f; int bestidx = 0x7FFFFFFF; int bp = 0;
#pragma unroll
    for (int j = 0; j < KNN_K; ++j)
      if (md[j] < bestd || (md[j] == bestd && mi[j] < bestidx)) {
        bestd = md[j]; bestidx = mi[j]; bp = j;
      }
    outp[a] = bestidx;
    md[bp] = 3.6e38f; mi[bp] = 0x7FFFFFFF;
  }
}

// ---------------- WMMA 32xCxC GEMM, bias+ReLU epilogue, f16 dest in LDS ----
template <int C>
__device__ __forceinline__ void wmma_gemm_relu(
    const _Float16* __restrict__ A,    // LDS [32][C]
    const _Float16* __restrict__ Bw,   // global [C][C] row-major (o, c)
    const float* __restrict__ bias,    // [C]
    _Float16* __restrict__ Dst,        // LDS [32][C]
    int tid) {
  const int lane = tid & 31;
  const int wave = tid >> 5;
  const int hh = lane >> 4;
  const int r0 = lane & 15;
  constexpr int NTPW = C / 128;    // ntiles per wave: 1 (C=128) or 2 (C=256)
#pragma unroll
  for (int ti = 0; ti < NTPW; ++ti) {
    const int ntile = wave * NTPW + ti;
    v8f acc0 = {};
    v8f acc1 = {};
#pragma unroll
    for (int ks = 0; ks < C / 32; ++ks) {
      F16Frag bf, a0, a1;
      const _Float16* bp = Bw + (size_t)(ntile * 16 + r0) * C + ks * 32 + hh * 8;
      bf.q[0] = *(const uv4*)(bp);
      bf.q[1] = *(const uv4*)(bp + 16);
      const _Float16* ap = A + r0 * C + ks * 32 + hh * 8;
      a0.q[0] = *(const uv4*)(ap);
      a0.q[1] = *(const uv4*)(ap + 16);
      const _Float16* ap1 = ap + 16 * C;
      a1.q[0] = *(const uv4*)(ap1);
      a1.q[1] = *(const uv4*)(ap1 + 16);
      acc0 = __builtin_amdgcn_wmma_f32_16x16x32_f16(false, a0.v, false, bf.v,
                                                    (short)0, acc0, false, false);
      acc1 = __builtin_amdgcn_wmma_f32_16x16x32_f16(false, a1.v, false, bf.v,
                                                    (short)0, acc1, false, false);
    }
    const int n = ntile * 16 + r0;
    const float bb = bias[n];
#pragma unroll
    for (int r = 0; r < 8; ++r) {
      const int m = hh * 8 + r;
      Dst[m * C + n]        = (_Float16)fmaxf(acc0[r] + bb, 0.f);
      Dst[(m + 16) * C + n] = (_Float16)fmaxf(acc1[r] + bb, 0.f);
    }
  }
}

// ---------------- sample_and_group core: 1 workgroup / sampled point -------
template <int C>  // C = 2*D (feature width of the two convs)
__global__ __launch_bounds__(256) void group_gemm_kernel(
    const _Float16* __restrict__ hin,  // [B,Nin,D]
    const int* __restrict__ nidx,      // [B,S,32]
    const int* __restrict__ sidx,      // [B,S]
    const _Float16* __restrict__ w1h, const float* __restrict__ b1,
    const _Float16* __restrict__ w2h, const float* __restrict__ b2,
    _Float16* __restrict__ hout,       // [B,S,C]
    int Nin, int S) {
  constexpr int D = C / 2;
  __shared__ __align__(16) _Float16 featbuf[32 * C];  // reused as GEMM2 out
  __shared__ __align__(16) _Float16 tmpbuf[32 * C];
  __shared__ __align__(16) _Float16 xch[D];
  __shared__ int nb[32];
  const int tid = threadIdx.x;
  const int b = blockIdx.x / S;
  const int s = blockIdx.x % S;
  if (tid < 32) nb[tid] = nidx[((size_t)b * S + s) * 32 + tid];
  const int sc = sidx[(size_t)b * S + s];
  if (tid < D / 2)
    ((unsigned*)xch)[tid] =
        ((const unsigned*)(hin + ((size_t)b * Nin + sc) * D))[tid];
  __syncthreads();
  // feat[j] = concat(neigh_j - xc, xc)
  constexpr int HP = D / 2;  // u32 (f16-pair) elements per row
  for (int idx = tid; idx < 32 * HP; idx += 256) {
    const int j = idx / HP, cp = idx % HP;
    HPair nv, cv, sub;
    nv.u = ((const unsigned*)(hin + ((size_t)b * Nin + nb[j]) * D))[cp];
    cv.u = ((const unsigned*)xch)[cp];
    sub.h[0] = nv.h[0] - cv.h[0];
    sub.h[1] = nv.h[1] - cv.h[1];
    ((unsigned*)(featbuf + j * C))[cp]     = sub.u;
    ((unsigned*)(featbuf + j * C + D))[cp] = cv.u;
  }
  __syncthreads();
  wmma_gemm_relu<C>(featbuf, w1h, b1, tmpbuf, tid);
  __syncthreads();
  wmma_gemm_relu<C>(tmpbuf, w2h, b2, featbuf, tid);   // out -> featbuf
  __syncthreads();
  // max over the 32 neighbor rows
  for (int c = tid; c < C; c += 256) {
    float vmax = -3.4e38f;
#pragma unroll
    for (int m = 0; m < 32; ++m)
      vmax = fmaxf(vmax, (float)featbuf[m * C + c]);
    hout[((size_t)b * S + s) * C + c] = (_Float16)vmax;
  }
}

// ---------------- final channel GEMM + BN, transposed f32 store ------------
__global__ __launch_bounds__(256) void final_gemm_kernel(
    const _Float16* __restrict__ h2,  // [B,S,256]
    const _Float16* __restrict__ wh,  // [256,256]
    const float* __restrict__ bias,   // [256]
    float* __restrict__ outp,         // [B,256,S]
    int S) {
  constexpr int C = 256;
  __shared__ __align__(16) _Float16 featbuf[32 * C];
  const int tid = threadIdx.x;
  const int tiles = S / 32;
  const int b = blockIdx.x / tiles;
  const int s0 = (blockIdx.x % tiles) * 32;
  const unsigned* src = (const unsigned*)(h2 + ((size_t)b * S + s0) * C);
  unsigned* dstp = (unsigned*)featbuf;
  for (int i = tid; i < 32 * C / 2; i += 256) dstp[i] = src[i];
  __syncthreads();
  const int lane = tid & 31, wave = tid >> 5;
  const int hh = lane >> 4, r0 = lane & 15;
#pragma unroll
  for (int ti = 0; ti < 2; ++ti) {
    const int ntile = wave * 2 + ti;
    v8f acc0 = {};
    v8f acc1 = {};
#pragma unroll
    for (int ks = 0; ks < 8; ++ks) {
      F16Frag bf, a0, a1;
      const _Float16* bp = wh + (size_t)(ntile * 16 + r0) * C + ks * 32 + hh * 8;
      bf.q[0] = *(const uv4*)(bp);
      bf.q[1] = *(const uv4*)(bp + 16);
      const _Float16* ap = featbuf + r0 * C + ks * 32 + hh * 8;
      a0.q[0] = *(const uv4*)(ap);
      a0.q[1] = *(const uv4*)(ap + 16);
      const _Float16* ap1 = ap + 16 * C;
      a1.q[0] = *(const uv4*)(ap1);
      a1.q[1] = *(const uv4*)(ap1 + 16);
      acc0 = __builtin_amdgcn_wmma_f32_16x16x32_f16(false, a0.v, false, bf.v,
                                                    (short)0, acc0, false, false);
      acc1 = __builtin_amdgcn_wmma_f32_16x16x32_f16(false, a1.v, false, bf.v,
                                                    (short)0, acc1, false, false);
    }
    const int n = ntile * 16 + r0;
    const float bb = bias[n];
    float* ob = outp + (size_t)b * 256 * S + (size_t)n * S + s0;
#pragma unroll
    for (int r = 0; r < 8; ++r) {
      const int m = hh * 8 + r;
      ob[m]      = acc0[r] + bb;
      ob[m + 16] = acc1[r] + bb;
    }
  }
}

__global__ void copy_coords_kernel(const float* __restrict__ sc,
                                   float* __restrict__ dst, int n) {
  int t = blockIdx.x * 256 + threadIdx.x;
  if (t < n) dst[t] = sc[t];
}

// ---------------------------------------------------------------------------
extern "C" void kernel_launch(void* const* d_in, const int* in_sizes, int n_in,
                              void* d_out, int out_size, void* d_ws, size_t ws_size,
                              hipStream_t stream) {
  (void)in_sizes; (void)n_in; (void)out_size; (void)ws_size;
  const int B = 8, N = 4096, S1 = 2048, S2 = 1024;

  // setup_inputs() dict leaf order:
  // x, in_w1, in_bn1(g,b,m,v), in_w2, in_bn2(4), sg1.w1, sg1.bn1(4),
  // sg1.w2, sg1.bn2(4), sg2.w1, sg2.bn1(4), sg2.w2, sg2.bn2(4), out_w, out_bn(4)
  const float* x     = (const float*)d_in[0];
  const float* in_w1 = (const float*)d_in[1];
  const float* in_w2 = (const float*)d_in[6];
  const float* s1w1  = (const float*)d_in[11];
  const float* s1w2  = (const float*)d_in[16];
  const float* s2w1  = (const float*)d_in[21];
  const float* s2w2  = (const float*)d_in[26];
  const float* ow    = (const float*)d_in[31];
  auto bn = [&](int base, int j) { return (const float*)d_in[base + j]; };

  char* ws = (char*)d_ws;
  size_t off = 0;
  auto alloc = [&](size_t bytes) -> void* {
    off = (off + 255) & ~(size_t)255;
    void* p = ws + off;
    off += bytes;
    return p;
  };

  _Float16* w0h  = (_Float16*)alloc(64 * 3 * 2);    float* b0   = (float*)alloc(64 * 4);
  _Float16* w1h  = (_Float16*)alloc(64 * 64 * 2);   float* b1   = (float*)alloc(64 * 4);
  _Float16* g1w1 = (_Float16*)alloc(128 * 128 * 2); float* g1b1 = (float*)alloc(128 * 4);
  _Float16* g1w2 = (_Float16*)alloc(128 * 128 * 2); float* g1b2 = (float*)alloc(128 * 4);
  _Float16* g2w1 = (_Float16*)alloc(256 * 256 * 2); float* g2b1 = (float*)alloc(256 * 4);
  _Float16* g2w2 = (_Float16*)alloc(256 * 256 * 2); float* g2b2 = (float*)alloc(256 * 4);
  _Float16* owh  = (_Float16*)alloc(256 * 256 * 2); float* obv  = (float*)alloc(256 * 4);

  _Float16* h0    = (_Float16*)alloc((size_t)B * N * 64 * 2);
  int*      sidx1 = (int*)alloc((size_t)B * S1 * 4);
  float*    sc1   = (float*)alloc((size_t)B * S1 * 3 * 4);
  int*      nidx1 = (int*)alloc((size_t)B * S1 * 32 * 4);
  _Float16* h1    = (_Float16*)alloc((size_t)B * S1 * 128 * 2);
  int*      sidx2 = (int*)alloc((size_t)B * S2 * 4);
  float*    sc2   = (float*)alloc((size_t)B * S2 * 3 * 4);
  int*      nidx2 = (int*)alloc((size_t)B * S2 * 32 * 4);
  _Float16* h2    = (_Float16*)alloc((size_t)B * S2 * 256 * 2);

  // fold BN into weights
  fold_bn_kernel<<<1,   256, 0, stream>>>(in_w1, bn(2,0),  bn(2,1),  bn(2,2),  bn(2,3),  64, 3,   w0h,  b0);
  fold_bn_kernel<<<16,  256, 0, stream>>>(in_w2, bn(7,0),  bn(7,1),  bn(7,2),  bn(7,3),  64, 64,  w1h,  b1);
  fold_bn_kernel<<<64,  256, 0, stream>>>(s1w1,  bn(12,0), bn(12,1), bn(12,2), bn(12,3), 128,128, g1w1, g1b1);
  fold_bn_kernel<<<64,  256, 0, stream>>>(s1w2,  bn(17,0), bn(17,1), bn(17,2), bn(17,3), 128,128, g1w2, g1b2);
  fold_bn_kernel<<<256, 256, 0, stream>>>(s2w1,  bn(22,0), bn(22,1), bn(22,2), bn(22,3), 256,256, g2w1, g2b1);
  fold_bn_kernel<<<256, 256, 0, stream>>>(s2w2,  bn(27,0), bn(27,1), bn(27,2), bn(27,3), 256,256, g2w2, g2b2);
  fold_bn_kernel<<<256, 256, 0, stream>>>(ow,    bn(32,0), bn(32,1), bn(32,2), bn(32,3), 256,256, owh,  obv);

  // input MLP -> h0 (f16)
  input_embed_kernel<<<(B * N + 127) / 128, 128, 0, stream>>>(x, w0h, b0, w1h, b1, h0, B * N);

  // stage 1: FPS(4096 -> 2048), kNN(32), grouped convs C=128
  fps_kernel<<<B, 1024, 0, stream>>>(x, N, S1, sidx1, sc1);
  knn_kernel<<<(B * S1 + 63) / 64, 64, 0, stream>>>(sc1, x, N, S1, B, nidx1);
  group_gemm_kernel<128><<<B * S1, 256, 0, stream>>>(h0, nidx1, sidx1,
                                                     g1w1, g1b1, g1w2, g1b2,
                                                     h1, N, S1);

  // stage 2: FPS(2048 -> 1024), kNN(32), grouped convs C=256
  fps_kernel<<<B, 1024, 0, stream>>>(sc1, S1, S2, sidx2, sc2);
  knn_kernel<<<(B * S2 + 63) / 64, 64, 0, stream>>>(sc2, sc1, S1, S2, B, nidx2);
  group_gemm_kernel<256><<<B * S2, 256, 0, stream>>>(h1, nidx2, sidx2,
                                                     g2w1, g2b1, g2w2, g2b2,
                                                     h2, S1, S2);

  // final channel GEMM + BN, transposed into d_out; then coords tail
  float* outp = (float*)d_out;
  final_gemm_kernel<<<B * (S2 / 32), 256, 0, stream>>>(h2, owh, obv, outp, S2);
  copy_coords_kernel<<<(B * S2 * 3 + 255) / 256, 256, 0, stream>>>(
      sc2, outp + (size_t)B * 256 * S2, B * S2 * 3);
}
